// VisionAttention_11854109737650
// MI455X (gfx1250) — compile-verified, matrix operand order checked by reading, not run on hardware
//
#include <hip/hip_runtime.h>

typedef __bf16 v16bf __attribute__((ext_vector_type(16)));
typedef __bf16 v8bf  __attribute__((ext_vector_type(8)));
typedef float  v8f   __attribute__((ext_vector_type(8)));
typedef float  v4f   __attribute__((ext_vector_type(4)));
typedef unsigned int u32x4 __attribute__((ext_vector_type(4)));
typedef int          i32x4 __attribute__((ext_vector_type(4)));
typedef int          i32x8 __attribute__((ext_vector_type(8)));

#define S_TOK  2048
#define DIM    1280
#define NH     16
#define HD     80
#define QKV_N  3840

#if defined(__gfx1250__) && __has_builtin(__builtin_amdgcn_tensor_load_to_lds) && \
    __has_builtin(__builtin_amdgcn_s_wait_tensorcnt)
#define USE_TDM 1
#else
#define USE_TDM 0
#endif

union FragBF { v16bf v; v8bf h[2]; };

__device__ __forceinline__ v8bf zero8() {
  v8bf z;
#pragma unroll
  for (int e = 0; e < 8; ++e) z[e] = (__bf16)0.0f;
  return z;
}

// Stage 8 contiguous elements into LDS as bf16 (16B-aligned src/dst).
__device__ __forceinline__ void stage8(const float* __restrict__ src, __bf16* dst) {
  v4f f0 = ((const v4f*)src)[0];
  v4f f1 = ((const v4f*)src)[1];
  v8bf b;
#pragma unroll
  for (int e = 0; e < 4; ++e) { b[e] = (__bf16)f0[e]; b[4 + e] = (__bf16)f1[e]; }
  *(v8bf*)dst = b;
}
__device__ __forceinline__ void stage8(const __bf16* __restrict__ src, __bf16* dst) {
  *(v8bf*)dst = *(const v8bf*)src;
}

// ---------------------------------------------------------------------------
// Tiled GEMM: C[M,N] = A[M,K] @ B[N,K]^T + bias[N]
// Block = 256 threads = 8 waves as 2(M) x 4(N); each wave owns a 32x32
// macro-tile (2x2 WMMA tiles). Block tile 64x128, K-step 64.
// -> 8 WMMAs + 16 ds_load_b128 per barrier pair, 4x fragment reuse.
// LDS rows padded to 72 elems (144B) so all fragment loads are b128.
// ---------------------------------------------------------------------------
template <typename AT, typename OT>
__global__ void gemm_bias_kernel(const AT* __restrict__ A, const float* __restrict__ B,
                                 const float* __restrict__ bias, OT* __restrict__ C,
                                 int M, int N, int K) {
  __shared__ __bf16 As[64][72];
  __shared__ __bf16 Bs[128][72];
  const int tid  = threadIdx.x;
  const int wid  = tid >> 5;
  const int lr   = tid & 15;
  const int half = (tid >> 4) & 1;
  const int wm = wid & 1, wn = wid >> 1;       // 2 x 4 wave grid
  const int m0 = blockIdx.y * 64;
  const int n0 = blockIdx.x * 128;

  v8f acc[2][2];
#pragma unroll
  for (int tm = 0; tm < 2; ++tm)
#pragma unroll
    for (int tn = 0; tn < 2; ++tn)
#pragma unroll
      for (int r = 0; r < 8; ++r) acc[tm][tn][r] = 0.0f;

  for (int k0 = 0; k0 < K; k0 += 64) {
    // Stage A tile 64x64: 512 chunks of 8, 2 per thread
#pragma unroll
    for (int i = 0; i < 2; ++i) {
      int idx = tid + i * 256;
      int r = idx >> 3, c8 = idx & 7;
      stage8(A + (size_t)(m0 + r) * K + k0 + c8 * 8, &As[r][c8 * 8]);
    }
    // Stage B tile 128x64: 1024 chunks of 8, 4 per thread
#pragma unroll
    for (int i = 0; i < 4; ++i) {
      int idx = tid + i * 256;
      int r = idx >> 3, c8 = idx & 7;
      stage8(B + (size_t)(n0 + r) * K + k0 + c8 * 8, &Bs[r][c8 * 8]);
    }
    if (k0 + 64 < K) {
      __builtin_prefetch(A + (size_t)(m0 + (tid >> 3)) * K + k0 + 64 + (tid & 7) * 8, 0, 0);
      __builtin_prefetch(B + (size_t)(n0 + (tid >> 1)) * K + k0 + 64, 0, 0);
    }
    __syncthreads();

#pragma unroll
    for (int ks = 0; ks < 2; ++ks) {           // two K=32 sub-steps
      FragBF af[2], bfr[2];
#pragma unroll
      for (int tm = 0; tm < 2; ++tm) {
        const v8bf* arow = (const v8bf*)&As[wm * 32 + tm * 16 + lr][0];  // 9 chunks/row
        af[tm].h[0] = arow[ks * 4 + half];
        af[tm].h[1] = arow[ks * 4 + 2 + half];
      }
#pragma unroll
      for (int tn = 0; tn < 2; ++tn) {
        const v8bf* brow = (const v8bf*)&Bs[wn * 32 + tn * 16 + lr][0];
        bfr[tn].h[0] = brow[ks * 4 + half];
        bfr[tn].h[1] = brow[ks * 4 + 2 + half];
      }
#pragma unroll
      for (int tm = 0; tm < 2; ++tm)
#pragma unroll
        for (int tn = 0; tn < 2; ++tn)
          acc[tm][tn] = __builtin_amdgcn_wmma_f32_16x16x32_bf16(
              false, af[tm].v, false, bfr[tn].v, (short)0, acc[tm][tn], false, false);
    }
    __syncthreads();
  }

#pragma unroll
  for (int tn = 0; tn < 2; ++tn) {
    int n = n0 + wn * 32 + tn * 16 + lr;
    float bv = bias[n];
#pragma unroll
    for (int tm = 0; tm < 2; ++tm)
#pragma unroll
      for (int r = 0; r < 8; ++r) {
        int m = m0 + wm * 32 + tm * 16 + r + half * 8;
        C[(size_t)m * N + n] = (OT)(acc[tm][tn][r] + bv);
      }
  }
}

// ---------------------------------------------------------------------------
// RoPE in place on q (slot 0) and k (slot 1) of the bf16 qkv buffer.
// ---------------------------------------------------------------------------
__global__ void rope_kernel(__bf16* __restrict__ qkv, const float* __restrict__ freqs) {
  int idx = blockIdx.x * blockDim.x + threadIdx.x;
  if (idx >= S_TOK * NH * (HD / 2)) return;
  int d2 = idx % (HD / 2);
  int h  = (idx / (HD / 2)) % NH;
  int s  = idx / ((HD / 2) * NH);
  float f = freqs[s * (HD / 2) + d2];
  float c = cosf(f), sn = sinf(f);
#pragma unroll
  for (int slot = 0; slot < 2; ++slot) {
    size_t base = (size_t)s * QKV_N + slot * DIM + h * HD;
    float x1 = (float)qkv[base + d2];
    float x2 = (float)qkv[base + d2 + HD / 2];
    qkv[base + d2]          = (__bf16)(x1 * c - x2 * sn);
    qkv[base + d2 + HD / 2] = (__bf16)(x2 * c + x1 * sn);
  }
}

// ---------------------------------------------------------------------------
// Segmented flash attention. Grid (S/64, NH), 128 threads = 4 waves.
// K tile loaded by TDM (tensor_load_to_lds) when available; V staged transposed
// so all WMMA fragments load as 16B ds_load_b128.
// ---------------------------------------------------------------------------
__global__ void attn_kernel(const __bf16* __restrict__ qkv, const int* __restrict__ cu,
                            __bf16* __restrict__ out) {
  __shared__ __bf16 Ks[32][HD];      // row stride 160B (16B-aligned)
  __shared__ __bf16 Vt[HD][32];      // transposed V: row stride 64B
  __shared__ __bf16 Ps[4][16][32];   // per-wave probs, row 64B

  const int tid  = threadIdx.x;
  const int wid  = tid >> 5;
  const int lr   = tid & 15;
  const int half = (tid >> 4) & 1;
  const int h    = blockIdx.y;
  const int qblk = blockIdx.x * 64;

  const int c1 = cu[1], c2 = cu[2], c3 = cu[3], c4 = cu[4];
  auto seg_of = [&](int i) -> int { return (i >= c1) + (i >= c2) + (i >= c3); };

  const int seg_lo = seg_of(qblk);
  const int seg_hi = seg_of(qblk + 63);
  const int kstart = (seg_lo == 0) ? cu[0] : (seg_lo == 1 ? c1 : (seg_lo == 2 ? c2 : c3));
  const int kend   = (seg_hi == 0) ? c1    : (seg_hi == 1 ? c2 : (seg_hi == 2 ? c3 : c4));

  // q fragments (3 HD-chunks of 32; chunk 2 zero-padded past 80), b128 global loads
  const int qrow = qblk + wid * 16 + lr;
  const __bf16* qp = qkv + (size_t)qrow * QKV_N + h * HD;
  FragBF aq[3];
#pragma unroll
  for (int c = 0; c < 2; ++c) {
    aq[c].h[0] = *(const v8bf*)(qp + c * 32 + half * 8);
    aq[c].h[1] = *(const v8bf*)(qp + c * 32 + 16 + half * 8);
  }
  aq[2].h[0] = *(const v8bf*)(qp + 64 + half * 8);
  aq[2].h[1] = zero8();

  int segq[8];
#pragma unroll
  for (int r = 0; r < 8; ++r) segq[r] = seg_of(qblk + wid * 16 + r + half * 8);

  float mrow[8], lsum[8];
  v8f o[5];
#pragma unroll
  for (int r = 0; r < 8; ++r) { mrow[r] = -3.0e38f; lsum[r] = 0.0f; }
#pragma unroll
  for (int f = 0; f < 5; ++f) {
#pragma unroll
    for (int r = 0; r < 8; ++r) o[f][r] = 0.0f;
  }

  const float scale = 0.11180339887498949f; // 1/sqrt(80)

  for (int j = kstart; j < kend; j += 32) {
    __syncthreads();

#if USE_TDM
    if (wid == 0) {
      // TDM 2D descriptor: tile 80x32 bf16 from K rows j..j+31; OOB rows zero-fill.
      unsigned lds_off = (unsigned)(uintptr_t)&Ks[0][0];
      unsigned long long ga =
          (unsigned long long)(uintptr_t)(qkv + (size_t)j * QKV_N + DIM + h * HD);
      unsigned nrows = (unsigned)(kend - j);
      u32x4 g0 = {1u,                                   // count=1
                  lds_off,                              // lds_addr
                  (unsigned)ga,                         // global_addr lo
                  (unsigned)((ga >> 32) & 0x1FFFFFFu) | (2u << 30)};  // addr hi | type=2
      i32x8 g1;
      g1[0] = (int)(1u << 16);                          // data_size=1 (2B)
      g1[1] = (int)((80u & 0xFFFFu) << 16);             // tensor_dim0[15:0]
      g1[2] = (int)((nrows & 0xFFFFu) << 16);           // dim0 hi=0 | tensor_dim1 lo
      g1[3] = (int)((nrows >> 16) | (80u << 16));       // dim1 hi | tile_dim0=80
      g1[4] = (int)32u;                                 // tile_dim1=32, tile_dim2=0
      g1[5] = (int)(unsigned)QKV_N;                     // tensor_dim0_stride lo
      g1[6] = 0;                                        // stride hi | dim1_stride
      g1[7] = 0;
      i32x4 z4 = {0, 0, 0, 0};
#if __clang_major__ >= 23
      i32x8 z8;
#pragma unroll
      for (int e = 0; e < 8; ++e) z8[e] = 0;
      __builtin_amdgcn_tensor_load_to_lds(g0, g1, z4, z4, z8, 0);
#else
      __builtin_amdgcn_tensor_load_to_lds(g0, g1, z4, z4, 0);
#endif
    }
#endif

    // Stage V transposed (and K manually when no TDM): 320 v8bf chunks / 128 threads
    for (int i = tid; i < 32 * (HD / 8); i += 128) {
      int r = i / (HD / 8), c8 = i % (HD / 8);
      int key = j + r;
      v8bf vv = zero8();
      if (key < kend)
        vv = *(const v8bf*)(qkv + (size_t)key * QKV_N + 2 * DIM + h * HD + c8 * 8);
#pragma unroll
      for (int e = 0; e < 8; ++e) Vt[c8 * 8 + e][r] = vv[e];
#if !USE_TDM
      v8bf kv = zero8();
      if (key < kend)
        kv = *(const v8bf*)(qkv + (size_t)key * QKV_N + DIM + h * HD + c8 * 8);
      *(v8bf*)&Ks[r][c8 * 8] = kv;
#endif
    }
#if USE_TDM
    if (wid == 0) __builtin_amdgcn_s_wait_tensorcnt(0);
#endif
    __syncthreads();

    // scores = q @ k^T (2 key halves x 3 HD chunks)
    v8f sc[2];
#pragma unroll
    for (int f = 0; f < 2; ++f) {
#pragma unroll
      for (int r = 0; r < 8; ++r) sc[f][r] = 0.0f;
      const __bf16* kp = &Ks[f * 16 + lr][0];
#pragma unroll
      for (int c = 0; c < 3; ++c) {
        FragBF bk;
        bk.h[0] = *(const v8bf*)(kp + c * 32 + half * 8);
        bk.h[1] = (c < 2) ? *(const v8bf*)(kp + c * 32 + 16 + half * 8) : zero8();
        sc[f] = __builtin_amdgcn_wmma_f32_16x16x32_bf16(false, aq[c].v, false, bk.v,
                                                        (short)0, sc[f], false, false);
      }
    }

    // segment mask + online softmax (row r+8*half; its 16 cols live in one half-wave)
    const int key0 = j + lr, key1 = j + 16 + lr;
    const int segk0 = (key0 < kend) ? seg_of(key0) : -1;
    const int segk1 = (key1 < kend) ? seg_of(key1) : -1;
#pragma unroll
    for (int r = 0; r < 8; ++r) {
      float s0 = sc[0][r] * scale;
      float s1 = sc[1][r] * scale;
      if (segk0 != segq[r]) s0 = -3.0e38f;
      if (segk1 != segq[r]) s1 = -3.0e38f;
      float rm = fmaxf(s0, s1);
#pragma unroll
      for (int off = 1; off < 16; off <<= 1)
        rm = fmaxf(rm, __shfl_xor(rm, off, 32));
      float mnew = fmaxf(mrow[r], rm);
      float corr = __expf(mrow[r] - mnew);
      float p0 = __expf(s0 - mnew);
      float p1 = __expf(s1 - mnew);
      float rs = p0 + p1;
#pragma unroll
      for (int off = 1; off < 16; off <<= 1)
        rs += __shfl_xor(rs, off, 32);
      lsum[r] = lsum[r] * corr + rs;
      mrow[r] = mnew;
#pragma unroll
      for (int f = 0; f < 5; ++f) o[f][r] *= corr;
      Ps[wid][r + half * 8][lr]      = (__bf16)p0;
      Ps[wid][r + half * 8][16 + lr] = (__bf16)p1;
    }
    __syncthreads();

    // O += P @ V : A = probs 16x32 (b128 loads), B = Vt rows (b128 loads)
    const v8bf* prow = (const v8bf*)&Ps[wid][lr][0];  // 4 chunks/row
    FragBF ap;
    ap.h[0] = prow[half];
    ap.h[1] = prow[2 + half];
#pragma unroll
    for (int f = 0; f < 5; ++f) {
      const v8bf* vrow = (const v8bf*)&Vt[f * 16 + lr][0];  // 4 chunks/row
      FragBF bv;
      bv.h[0] = vrow[half];
      bv.h[1] = vrow[2 + half];
      o[f] = __builtin_amdgcn_wmma_f32_16x16x32_bf16(false, ap.v, false, bv.v,
                                                     (short)0, o[f], false, false);
    }
  }

  // Normalize and write (S, DIM), col = h*HD + hd
#pragma unroll
  for (int f = 0; f < 5; ++f) {
#pragma unroll
    for (int r = 0; r < 8; ++r) {
      int row = qblk + wid * 16 + r + half * 8;
      int col = h * HD + f * 16 + lr;
      out[(size_t)row * DIM + col] = (__bf16)(o[f][r] / lsum[r]);
    }
  }
}

// ---------------------------------------------------------------------------
extern "C" void kernel_launch(void* const* d_in, const int* in_sizes, int n_in,
                              void* d_out, int out_size, void* d_ws, size_t ws_size,
                              hipStream_t stream) {
  (void)in_sizes; (void)n_in; (void)out_size; (void)ws_size;
  const float* hidden = (const float*)d_in[0];
  const int*   cu     = (const int*)d_in[1];
  const float* rot    = (const float*)d_in[2];
  const float* qkv_w  = (const float*)d_in[3];
  const float* qkv_b  = (const float*)d_in[4];
  const float* proj_w = (const float*)d_in[5];
  const float* proj_b = (const float*)d_in[6];
  float* out = (float*)d_out;

  __bf16* ws_qkv  = (__bf16*)d_ws;  // 2048*3840 bf16
  __bf16* ws_attn = (__bf16*)((char*)d_ws +
                              (size_t)S_TOK * QKV_N * sizeof(unsigned short));

  gemm_bias_kernel<float, __bf16>
      <<<dim3(QKV_N / 128, S_TOK / 64), 256, 0, stream>>>(
          hidden, qkv_w, qkv_b, ws_qkv, S_TOK, QKV_N, DIM);

  rope_kernel<<<(S_TOK * NH * (HD / 2) + 255) / 256, 256, 0, stream>>>(ws_qkv, rot);

  attn_kernel<<<dim3(S_TOK / 64, NH), 128, 0, stream>>>(ws_qkv, cu, ws_attn);

  gemm_bias_kernel<__bf16, float>
      <<<dim3(DIM / 128, S_TOK / 64), 256, 0, stream>>>(
          ws_attn, proj_w, proj_b, out, S_TOK, DIM, DIM);
}
